// Multi_Encoding_net_67723044323999
// MI455X (gfx1250) — compile-verified, matrix operand order checked by reading, not run on hardware
//
#include <hip/hip_runtime.h>
#include <hip/hip_bf16.h>
#include <stdint.h>
#include <stddef.h>

// ---------------------------------------------------------------------------
// MI455X (gfx1250) implementation. wave32, WMMA 16x16x32 bf16 (f32 accum),
// 2x2 register macro-tiles, async global->LDS staging (ASYNCcnt).
// ---------------------------------------------------------------------------

typedef __attribute__((ext_vector_type(16))) __bf16 v16bf;
typedef __attribute__((ext_vector_type(8)))  float  v8f;

#define DEV static __device__ __forceinline__

union Frag  { v16bf v; uint4 q[2]; };
union Pack8 { __bf16 h[8]; uint4 q; };

// Problem constants (match reference)
constexpr int BB   = 4;
constexpr int NPTS = 8192;
constexpr int NSEL = 1024;
constexpr int DF   = 64;
constexpr int GD   = 512;
constexpr int CCAT = 1280;           // 128+256+384 + 512

// Workspace layout (bytes)
constexpr size_t OFF_IDX0 = 0;                                  // 4096*16 int
constexpr size_t OFF_IDX1 = OFF_IDX0 + (size_t)4096*16*4;       // 4096*32 int
constexpr size_t OFF_IDX2 = OFF_IDX1 + (size_t)4096*32*4;       // 4096*64 int
constexpr size_t OFF_XBFT = OFF_IDX2 + (size_t)4096*64*4;       // [4096][1280] bf16
constexpr size_t OFF_H1T  = OFF_XBFT + (size_t)4096*1280*2;     // [4096][512]  bf16
constexpr size_t OFF_WBF  = OFF_H1T  + (size_t)4096*512*2;      // bf16 weights
constexpr size_t WBF_ELEMS = 956416;
constexpr size_t OFF_SS   = OFF_WBF + WBF_ELEMS*2;              // 3200 floats

constexpr int W1OFF = 235520;   // 512x1280 bf16 head weights
constexpr int W2OFF = 890880;   // 128x512

DEV v8f wmma_bf16(v16bf a, v16bf b, v8f c) {
  return __builtin_amdgcn_wmma_f32_16x16x32_bf16(false, a, false, b, (short)0, c,
                                                 false, false);
}

// One fragment half-row: 16B at p plus 16B at p+16 elements (k 0-7, 16-23).
// p must already include row*stride + kc*32 + (lane&16 ? 8 : 0).
DEV v16bf load_frag(const __bf16* p) {
  Frag f;
  f.q[0] = *(const uint4*)(p);
  f.q[1] = *(const uint4*)(p + 16);
  return f.v;
}

DEV float lrelu(float x) { return fmaxf(x, 0.2f * x); }

// 2x2 macro-tile K-loop: 4 fragment loads feed 4 WMMAs per k-step.
DEV void gemm2x2(const __bf16* a0, const __bf16* a1, const __bf16* b0,
                 const __bf16* b1, int CinP, v8f& c00, v8f& c01, v8f& c10,
                 v8f& c11) {
  for (int kc = 0; kc < CinP; kc += 32) {
    v16bf af0 = load_frag(a0 + kc);
    v16bf af1 = load_frag(a1 + kc);
    v16bf bf0 = load_frag(b0 + kc);
    v16bf bf1 = load_frag(b1 + kc);
    c00 = wmma_bf16(af0, bf0, c00);
    c01 = wmma_bf16(af0, bf1, c01);
    c10 = wmma_bf16(af1, bf0, c10);
    c11 = wmma_bf16(af1, bf1, c11);
  }
}

// ---------------------------------------------------------------------------
// 1) Ball query, all 3 scales in one pass.  16 blocks x 256 threads.
//    xyz staged into LDS with CDNA5 async global->LDS B128 copies.
// ---------------------------------------------------------------------------
__global__ __launch_bounds__(256) void ball_query_kernel(
    const float* __restrict__ xyz, const int* __restrict__ sel,
    const float* __restrict__ radius,
    int* __restrict__ idx0, int* __restrict__ idx1, int* __restrict__ idx2) {
  __shared__ __align__(16) float sxyz[NPTS * 3];

  const int b  = blockIdx.x >> 2;
  const int sc = blockIdx.x & 3;
  const int tid = threadIdx.x;

  // Async bulk copy: 8192*3 floats = 6144 x 16B chunks, ASYNCcnt-tracked.
  const size_t base = (size_t)b * NPTS * 3;
  for (int e = tid; e < (NPTS * 3) / 4; e += 256) {
    const float* g = xyz + base + (size_t)e * 4;
    const uint32_t l = (uint32_t)(uintptr_t)(const void*)&sxyz[e * 4];
    asm volatile("global_load_async_to_lds_b128 %0, %1, off"
                 :: "v"(l), "v"(g) : "memory");
  }
  asm volatile("s_wait_asynccnt 0x0" ::: "memory");
  __syncthreads();

  const int s = sc * 256 + tid;
  const int cs = b * NSEL + s;
  const int ci = sel[cs];
  const float cx = sxyz[ci * 3 + 0];
  const float cy = sxyz[ci * 3 + 1];
  const float cz = sxyz[ci * 3 + 2];

  const float r = radius[0];
  const float r0 = r * r;
  const float r1 = (r / 1.5f) * (r / 1.5f);
  const float r2 = (r / 2.25f) * (r / 2.25f);

  int c0 = 0, c1 = 0, c2 = 0;
  int f0 = 0, f1 = 0, f2 = 0;
  int* o0 = idx0 + (size_t)cs * 16;
  int* o1 = idx1 + (size_t)cs * 32;
  int* o2 = idx2 + (size_t)cs * 64;

  for (int j = 0; j < NPTS; ++j) {
    const float dx = sxyz[j * 3 + 0] - cx;
    const float dy = sxyz[j * 3 + 1] - cy;
    const float dz = sxyz[j * 3 + 2] - cz;
    const float d2 = dx * dx + dy * dy + dz * dz;
    if (d2 < r0) {
      if (c0 < 16) { o0[c0] = j; if (c0 == 0) f0 = j; ++c0; }
      if (d2 < r1) {
        if (c1 < 32) { o1[c1] = j; if (c1 == 0) f1 = j; ++c1; }
        if (d2 < r2) {
          if (c2 < 64) { o2[c2] = j; if (c2 == 0) f2 = j; ++c2; }
        }
      }
      if (c0 == 16 && c1 == 32 && c2 == 64) break;
    }
  }
  const int p0 = (c0 > 0) ? f0 : 0;
  const int p1 = (c1 > 0) ? f1 : 0;
  const int p2 = (c2 > 0) ? f2 : 0;
  for (int k = c0; k < 16; ++k) o0[k] = p0;
  for (int k = c1; k < 32; ++k) o1[k] = p1;
  for (int k = c2; k < 64; ++k) o2[k] = p2;
}

// ---------------------------------------------------------------------------
// 2) Weight prep: fp32 (Cout x Cin) -> bf16 (Cout x CinP) zero-padded,
//    and fold bias/gamma/beta into per-channel scale/shift.
// ---------------------------------------------------------------------------
__global__ __launch_bounds__(256) void convert_w_kernel(
    const float* __restrict__ w, __bf16* __restrict__ dst,
    int Cout, int Cin, int CinP) {
  const int e = blockIdx.x * 256 + threadIdx.x;
  const int total = Cout * CinP;
  if (e >= total) return;
  const int o = e / CinP, c = e % CinP;
  dst[e] = (__bf16)((c < Cin) ? w[o * Cin + c] : 0.0f);
}

__global__ __launch_bounds__(256) void make_ss_kernel(
    const float* __restrict__ bias, const float* __restrict__ beta,
    const float* __restrict__ gamma, float* __restrict__ dst, int C) {
  const int e = blockIdx.x * 256 + threadIdx.x;
  if (e >= C) return;
  dst[e]     = gamma[e];
  dst[C + e] = gamma[e] * bias[e] + beta[e];
}

// ---------------------------------------------------------------------------
// 3) Broadcast global features into concat buffer channels [768,1280).
// ---------------------------------------------------------------------------
__global__ __launch_bounds__(256) void gf_broadcast_kernel(
    const float* __restrict__ gf, __bf16* __restrict__ xbf_t) {
  const int e = blockIdx.x * 256 + threadIdx.x;   // B*S*GD = 2^21
  const int b = e >> 19;
  const int rem = e & ((1 << 19) - 1);
  const int s = rem >> 9;
  const int c = rem & 511;
  xbf_t[(size_t)(b * NSEL + s) * CCAT + 768 + c] = (__bf16)gf[b * GD + c];
}

// ---------------------------------------------------------------------------
// 4) Fused gather + 3-layer MLP (WMMA bf16) + max-over-K, one scale.
//    64 columns per workgroup, activations column-major in LDS so every
//    fragment is two contiguous ds_load_b128 per lane.
// ---------------------------------------------------------------------------
template <int SCALE>
__global__ __launch_bounds__(256) void mlp_scale_kernel(
    const float* __restrict__ xyz, const float* __restrict__ points,
    const int* __restrict__ sel, const int* __restrict__ idxbuf,
    const __bf16* __restrict__ wbf, const float* __restrict__ ss,
    __bf16* __restrict__ xbf_t) {
  constexpr int KNB = (SCALE == 0) ? 16 : (SCALE == 1) ? 32 : 64;
  constexpr int CT  = 64 / KNB;
  constexpr int CH0 = 96;  // 70 real channels padded to 96
  constexpr int CH1 = (SCALE == 0) ? 64 : 128;
  constexpr int CH2 = (SCALE == 0) ? 128 : (SCALE == 1) ? 128 : 256;
  constexpr int CH3 = (SCALE == 0) ? 128 : (SCALE == 1) ? 256 : 384;
  constexpr int CHOFF = (SCALE == 0) ? 0 : (SCALE == 1) ? 128 : 384;
  constexpr int WOFF[3] = {
      SCALE == 0 ? 0     : SCALE == 1 ? 30720 : 92160,
      SCALE == 0 ? 6144  : SCALE == 1 ? 43008 : 104448,
      SCALE == 0 ? 14336 : SCALE == 1 ? 59392 : 137216};
  constexpr int SSOFF[3] = {
      SCALE == 0 ? 0   : SCALE == 1 ? 640  : 1664,
      SCALE == 0 ? 128 : SCALE == 1 ? 896  : 1920,
      SCALE == 0 ? 384 : SCALE == 1 ? 1152 : 2432};

  __shared__ __align__(16) __bf16 sbuf0[64 * CH3];
  __shared__ __align__(16) __bf16 sbuf1[64 * CH3];

  const int tid  = threadIdx.x;
  const int lane = tid & 31;
  const int wid  = tid >> 5;
  const int blocksPerB = NSEL / CT;
  const int b  = blockIdx.x / blocksPerB;
  const int s0 = (blockIdx.x % blocksPerB) * CT;

  // ---- Gather stage: g = [p-ctr | p | feat], column-major, zero-padded ----
  {
    const int j  = tid & 63;          // column within tile
    const int cg = tid >> 6;          // 4 threads per column
    const int cc = j / KNB;
    const int k  = j % KNB;
    const int sidx = b * NSEL + (s0 + cc);
    const int i  = idxbuf[(size_t)sidx * KNB + k];
    const int ci = sel[sidx];
    const float* pp = xyz + ((size_t)b * NPTS + i) * 3;
    const float* pc = xyz + ((size_t)b * NPTS + ci) * 3;
    const float px = pp[0], py = pp[1], pz = pp[2];
    const float cx = pc[0], cy = pc[1], cz = pc[2];
    __bf16* col = sbuf0 + j * CH0;
    for (int c = cg; c < CH0; c += 4) {
      float v;
      if      (c == 0) v = px - cx;
      else if (c == 1) v = py - cy;
      else if (c == 2) v = pz - cz;
      else if (c == 3) v = px;
      else if (c == 4) v = py;
      else if (c == 5) v = pz;
      else if (c < 70) v = points[((size_t)b * DF + (c - 6)) * NPTS + i];
      else             v = 0.0f;
      col[c] = (__bf16)v;
    }
  }
  __syncthreads();

  // ---- One WMMA layer, 2x2 macro-tiles: out[col][o]=lrelu(sc*(W in)+sh) ----
  auto do_layer = [&](int CinP, int Cout, const __bf16* wl, const float* ssl,
                      const __bf16* in, __bf16* out) {
    const int macros = (Cout >> 5) * 2;          // (Cout/32) m-macros x 2 n-macros
    const int khalf = (lane & 16) ? 8 : 0;
    const int ohalf = (lane & 16) ? 8 : 0;
    for (int t = wid; t < macros; t += 8) {
      const int mm = t >> 1, nn = t & 1;
      const int row0 = mm * 32 + (lane & 15);
      const int col0 = nn * 32 + (lane & 15);
      const __bf16* a0 = wl + (size_t)row0 * CinP + khalf;
      const __bf16* a1 = a0 + (size_t)16 * CinP;
      const __bf16* b0 = in + (size_t)col0 * CinP + khalf;
      const __bf16* b1 = b0 + (size_t)16 * CinP;
      v8f c00 = {0,0,0,0,0,0,0,0}, c01 = c00, c10 = c00, c11 = c00;
      gemm2x2(a0, a1, b0, b1, CinP, c00, c01, c10, c11);
      const float* scv = ssl;
      const float* shv = ssl + Cout;
#pragma unroll
      for (int i = 0; i < 2; ++i) {
        const int obase = mm * 32 + i * 16 + ohalf;
        v8f ci0 = (i == 0) ? c00 : c10;
        v8f ci1 = (i == 0) ? c01 : c11;
#pragma unroll
        for (int jn = 0; jn < 2; ++jn) {
          const int col = nn * 32 + jn * 16 + (lane & 15);
          v8f acc = (jn == 0) ? ci0 : ci1;
          Pack8 pk;
#pragma unroll
          for (int r = 0; r < 8; ++r) {
            float y = fmaf(scv[obase + r], acc[r], shv[obase + r]);
            pk.h[r] = (__bf16)lrelu(y);
          }
          *(uint4*)(out + (size_t)col * Cout + obase) = pk.q;
        }
      }
    }
  };

  do_layer(CH0, CH1, wbf + WOFF[0], ss + SSOFF[0], sbuf0, sbuf1);
  __syncthreads();
  do_layer(CH1, CH2, wbf + WOFF[1], ss + SSOFF[1], sbuf1, sbuf0);
  __syncthreads();
  do_layer(CH2, CH3, wbf + WOFF[2], ss + SSOFF[2], sbuf0, sbuf1);
  __syncthreads();

  // ---- Max over K neighbors, write transposed concat buffer ----
  for (int p = tid; p < CH3 * CT; p += 256) {
    const int o  = p % CH3;
    const int cc = p / CH3;
    const __bf16* colbase = sbuf1 + (size_t)cc * KNB * CH3 + o;
    float m = -3.0e38f;
    for (int k = 0; k < KNB; ++k) {
      float v = (float)colbase[(size_t)k * CH3];
      m = fmaxf(m, v);
    }
    xbf_t[(size_t)(b * NSEL + s0 + cc) * CCAT + CHOFF + o] = (__bf16)m;
  }
}

// ---------------------------------------------------------------------------
// 5) Head GEMM 1: 1280 -> 512, lrelu.  2x2 macro-tiles straight from L2.
// ---------------------------------------------------------------------------
__global__ __launch_bounds__(256) void head1_kernel(
    const __bf16* __restrict__ xbf_t, const __bf16* __restrict__ w1,
    const float* __restrict__ b1, __bf16* __restrict__ h1t) {
  const int tid = threadIdx.x, lane = tid & 31, wid = tid >> 5;
  const int colB = blockIdx.x * 64;
  const int khalf = (lane & 16) ? 8 : 0;
  for (int t = wid; t < 32; t += 8) {            // 16 m-macros x 2 n-macros
    const int mm = t >> 1, nn = t & 1;
    const int row0 = mm * 32 + (lane & 15);
    const int col0 = colB + nn * 32 + (lane & 15);
    const __bf16* a0 = w1 + (size_t)row0 * CCAT + khalf;
    const __bf16* a1 = a0 + (size_t)16 * CCAT;
    const __bf16* bp0 = xbf_t + (size_t)col0 * CCAT + khalf;
    const __bf16* bp1 = bp0 + (size_t)16 * CCAT;
    v8f c00 = {0,0,0,0,0,0,0,0}, c01 = c00, c10 = c00, c11 = c00;
    gemm2x2(a0, a1, bp0, bp1, CCAT, c00, c01, c10, c11);
#pragma unroll
    for (int i = 0; i < 2; ++i) {
      const int obase = mm * 32 + i * 16 + ((lane & 16) ? 8 : 0);
#pragma unroll
      for (int jn = 0; jn < 2; ++jn) {
        const int col = colB + nn * 32 + jn * 16 + (lane & 15);
        v8f acc = (i == 0) ? ((jn == 0) ? c00 : c01) : ((jn == 0) ? c10 : c11);
        Pack8 pk;
#pragma unroll
        for (int r = 0; r < 8; ++r)
          pk.h[r] = (__bf16)lrelu(acc[r] + b1[obase + r]);
        *(uint4*)(h1t + (size_t)col * 512 + obase) = pk.q;
      }
    }
  }
}

// ---------------------------------------------------------------------------
// 6) Head GEMM 2 (512->128, lrelu) fused with final 128->6 linear and the
//    (2,3) min/max epilogue.  Output fp32 (B,S,2,3).
// ---------------------------------------------------------------------------
__global__ __launch_bounds__(256) void head2_kernel(
    const __bf16* __restrict__ h1t, const __bf16* __restrict__ w2,
    const float* __restrict__ b2, const float* __restrict__ w3,
    const float* __restrict__ b3, float* __restrict__ out) {
  __shared__ __align__(16) __bf16 sH[64 * 128];
  const int tid = threadIdx.x, lane = tid & 31, wid = tid >> 5;
  const int colB = blockIdx.x * 64;
  const int khalf = (lane & 16) ? 8 : 0;
  for (int t = wid; t < 8; t += 8) {             // 4 m-macros x 2 n-macros
    const int mm = t >> 1, nn = t & 1;
    const int row0 = mm * 32 + (lane & 15);
    const int col0 = nn * 32 + (lane & 15);
    const __bf16* a0 = w2 + (size_t)row0 * 512 + khalf;
    const __bf16* a1 = a0 + (size_t)16 * 512;
    const __bf16* bp0 = h1t + (size_t)(colB + col0) * 512 + khalf;
    const __bf16* bp1 = bp0 + (size_t)16 * 512;
    v8f c00 = {0,0,0,0,0,0,0,0}, c01 = c00, c10 = c00, c11 = c00;
    gemm2x2(a0, a1, bp0, bp1, 512, c00, c01, c10, c11);
#pragma unroll
    for (int i = 0; i < 2; ++i) {
      const int obase = mm * 32 + i * 16 + ((lane & 16) ? 8 : 0);
#pragma unroll
      for (int jn = 0; jn < 2; ++jn) {
        const int col = nn * 32 + jn * 16 + (lane & 15);
        v8f acc = (i == 0) ? ((jn == 0) ? c00 : c01) : ((jn == 0) ? c10 : c11);
        Pack8 pk;
#pragma unroll
        for (int r = 0; r < 8; ++r)
          pk.h[r] = (__bf16)lrelu(acc[r] + b2[obase + r]);
        *(uint4*)(sH + (size_t)col * 128 + obase) = pk.q;
      }
    }
  }
  __syncthreads();

  if (tid < 64) {
    const int col = colB + tid;
    const int b = col >> 10;
    const int s = col & 1023;
    float a6[6];
#pragma unroll
    for (int o = 0; o < 6; ++o) a6[o] = b3[o];
    const __bf16* hx = sH + (size_t)tid * 128;
    for (int c = 0; c < 128; ++c) {
      const float xv = (float)hx[c];
#pragma unroll
      for (int o = 0; o < 6; ++o) a6[o] = fmaf(w3[o * 128 + c], xv, a6[o]);
    }
    float* po = out + (size_t)(b * NSEL + s) * 6;
#pragma unroll
    for (int d = 0; d < 3; ++d) {
      po[d]     = fminf(a6[d], a6[3 + d]);
      po[3 + d] = fmaxf(a6[d], a6[3 + d]);
    }
  }
}

// ---------------------------------------------------------------------------
// Host launcher
// ---------------------------------------------------------------------------
extern "C" void kernel_launch(void* const* d_in, const int* in_sizes, int n_in,
                              void* d_out, int out_size, void* d_ws,
                              size_t ws_size, hipStream_t stream) {
  (void)in_sizes; (void)n_in; (void)out_size; (void)ws_size;

  const float* xyz    = (const float*)d_in[0];
  const float* points = (const float*)d_in[1];
  const float* gf     = (const float*)d_in[2];
  const int*   sel    = (const int*)d_in[3];
  const float* radius = (const float*)d_in[4];
  // params flattened: blocks (b,beta,gamma,w per layer), then head (b,w)
  const float* b1 = (const float*)d_in[41];
  const float* b2 = (const float*)d_in[43];
  const float* b3 = (const float*)d_in[45];
  const float* w3 = (const float*)d_in[46];

  char* ws = (char*)d_ws;
  int*    idx0  = (int*)(ws + OFF_IDX0);
  int*    idx1  = (int*)(ws + OFF_IDX1);
  int*    idx2  = (int*)(ws + OFF_IDX2);
  __bf16* xbf_t = (__bf16*)(ws + OFF_XBFT);
  __bf16* h1t   = (__bf16*)(ws + OFF_H1T);
  __bf16* wbf   = (__bf16*)(ws + OFF_WBF);
  float*  ssb   = (float*)(ws + OFF_SS);

  struct WJob { int i, co, ci, cip, off; };
  const WJob wj[11] = {
      {8, 64, 70, 96, 0},        {12, 128, 64, 64, 6144},
      {16, 128, 128, 128, 14336},{20, 128, 70, 96, 30720},
      {24, 128, 128, 128, 43008},{28, 256, 128, 128, 59392},
      {32, 128, 70, 96, 92160},  {36, 256, 128, 128, 104448},
      {40, 384, 256, 256, 137216},{42, 512, 1280, 1280, W1OFF},
      {44, 128, 512, 512, W2OFF}};
  for (int j = 0; j < 11; ++j) {
    const int total = wj[j].co * wj[j].cip;
    convert_w_kernel<<<(total + 255) / 256, 256, 0, stream>>>(
        (const float*)d_in[wj[j].i], wbf + wj[j].off, wj[j].co, wj[j].ci,
        wj[j].cip);
  }
  struct SJob { int ib, ibeta, igam, C, off; };
  const SJob sj[9] = {
      {5, 6, 7, 64, 0},     {9, 10, 11, 128, 128},  {13, 14, 15, 128, 384},
      {17, 18, 19, 128, 640},{21, 22, 23, 128, 896},{25, 26, 27, 256, 1152},
      {29, 30, 31, 128, 1664},{33, 34, 35, 256, 1920},{37, 38, 39, 384, 2432}};
  for (int j = 0; j < 9; ++j) {
    make_ss_kernel<<<(sj[j].C + 255) / 256, 256, 0, stream>>>(
        (const float*)d_in[sj[j].ib], (const float*)d_in[sj[j].ibeta],
        (const float*)d_in[sj[j].igam], ssb + sj[j].off, sj[j].C);
  }

  gf_broadcast_kernel<<<(BB * NSEL * GD) / 256, 256, 0, stream>>>(gf, xbf_t);

  ball_query_kernel<<<BB * 4, 256, 0, stream>>>(xyz, sel, radius, idx0, idx1,
                                                idx2);

  mlp_scale_kernel<0><<<BB * (NSEL / 4), 256, 0, stream>>>(
      xyz, points, sel, idx0, wbf, ssb, xbf_t);
  mlp_scale_kernel<1><<<BB * (NSEL / 2), 256, 0, stream>>>(
      xyz, points, sel, idx1, wbf, ssb, xbf_t);
  mlp_scale_kernel<2><<<BB * NSEL, 256, 0, stream>>>(
      xyz, points, sel, idx2, wbf, ssb, xbf_t);

  head1_kernel<<<(BB * NSEL) / 64, 256, 0, stream>>>(xbf_t, wbf + W1OFF,
                                                     (const float*)d_in[41],
                                                     h1t);
  head2_kernel<<<(BB * NSEL) / 64, 256, 0, stream>>>(h1t, wbf + W2OFF, b2, w3,
                                                     b3, (float*)d_out);
  (void)b1;
}
